// convbnrelu_ccnet_SiSRUconv_43078521979475
// MI455X (gfx1250) — matrix operand order repaired and checked by reading, not compiled, not run on hardware
//
#include <hip/hip_runtime.h>

#define BB 2
#define CC 16
#define DD 16
#define HH 128
#define WW 128
#define HW (HH*WW)
#define EPS 1e-5f

typedef __attribute__((ext_vector_type(16))) __bf16 bf16x16;
typedef __attribute__((ext_vector_type(8)))  float  v8f;

union FragAB { bf16x16 v; unsigned u[8]; };

__device__ inline v8f wmma_bf16(const FragAB& A, const FragAB& B, v8f C) {
  return __builtin_amdgcn_wmma_f32_16x16x32_bf16(false, A.v, false, B.v,
                                                 (short)0, C, false, false);
}

// ---------------------------------------------------------------------------
// Fold BN into weights, reorder to B[co][K] bf16 (K-order: (..kh,kw)*16+ci)
// ---------------------------------------------------------------------------
__global__ __launch_bounds__(256) void k_prep(
    const float* __restrict__ w_pre, const float* __restrict__ gp,
    const float* __restrict__ bp, const float* __restrict__ mp,
    const float* __restrict__ vp,
    const float* __restrict__ w_gate, const float* __restrict__ gg,
    const float* __restrict__ bg, const float* __restrict__ mg,
    const float* __restrict__ vg,
    __bf16* __restrict__ BwPre, float* __restrict__ biasPre,
    __bf16* __restrict__ BwGate, float* __restrict__ biasGate)
{
  int tid = threadIdx.x;
  for (int i = tid; i < 16*160; i += 256) {
    int co = i / 160, k = i % 160;
    float val = 0.f;
    if (k < 144) {
      int kh = k / 48, r = k % 48, kwid = r / 16, ci = r % 16;
      float s = gp[co] * rsqrtf(vp[co] + EPS);
      val = w_pre[(co*16 + ci)*9 + kh*3 + kwid] * s;
    }
    BwPre[co*160 + k] = (__bf16)val;
  }
  if (tid < 16) {
    float s = gp[tid] * rsqrtf(vp[tid] + EPS);
    biasPre[tid] = bp[tid] - mp[tid]*s;
  }
  for (int i = tid; i < 64*448; i += 256) {
    int co = i / 448, k = i % 448;
    float val = 0.f;
    if (k < 432) {
      int idx = k >> 4, ci = k & 15;
      int kd = idx / 9, r = idx % 9, kh = r / 3, kwid = r % 3;
      float s = gg[co] * rsqrtf(vg[co] + EPS);
      val = w_gate[((co*16 + ci)*3 + kd)*9 + kh*3 + kwid] * s;
    }
    BwGate[co*448 + k] = (__bf16)val;
  }
  if (tid < 64) {
    float s = gg[tid] * rsqrtf(vg[tid] + EPS);
    biasGate[tid] = bg[tid] - mg[tid]*s;
  }
}

// ---------------------------------------------------------------------------
// 1x3x3 conv + BN + ReLU :  M=128 (w), N=16 (co), K=160 (padded 144)
// ---------------------------------------------------------------------------
__global__ __launch_bounds__(256) void k_preconv(
    const float* __restrict__ x, const __bf16* __restrict__ BwPre,
    const float* __restrict__ biasPre, float* __restrict__ x1)
{
  __shared__ __bf16 xl[3][132][16];
  __shared__ __bf16 Bl[16*160];
  int tid = threadIdx.x;
  int h0 = blockIdx.x % HH;
  int d  = (blockIdx.x / HH) % DD;
  int b  = blockIdx.x / (HH*DD);

  for (int i = tid; i < 3*130*16; i += 256) {
    int kh = i / (130*16), r = i % (130*16), wsi = r >> 4, ci = r & 15;
    int hh = h0 - 1 + kh, ww = wsi - 1;
    float val = 0.f;
    if (hh >= 0 && hh < HH && ww >= 0 && ww < WW)
      val = x[((b*CC + ci)*DD + d)*HW + hh*WW + ww];
    xl[kh][wsi][ci] = (__bf16)val;
  }
  for (int i = tid; i < 16*160/2; i += 256)
    ((unsigned*)Bl)[i] = ((const unsigned*)BwPre)[i];
  __syncthreads();

  int lane = tid & 31, wave = tid >> 5;
  int m = lane & 15, kset = lane >> 4;
  int wbase = wave * 16;
  v8f acc; for (int i = 0; i < 8; ++i) acc[i] = 0.f;

  for (int kb = 0; kb < 5; ++kb) {
    FragAB A, Bf;
#pragma unroll
    for (int t = 0; t < 8; ++t) {
      int k = kb*32 + (t < 4 ? (kset*8 + 2*t) : (16 + kset*8 + 2*(t-4)));
      int ci = k & 15, idx = k >> 4;
      unsigned dv = 0u;
      if (idx < 9) {
        int kh = idx / 3, kwid = idx % 3;
        dv = *(const unsigned*)&xl[kh][wbase + m + kwid][ci];
      }
      A.u[t]  = dv;
      Bf.u[t] = *(const unsigned*)&Bl[m*160 + k];
    }
    acc = wmma_bf16(A, Bf, acc);
  }
  float bias = biasPre[m];
#pragma unroll
  for (int r = 0; r < 8; ++r) {
    int wpix = wbase + r + 8*kset;
    float val = acc[r] + bias;
    x1[((b*CC + m)*DD + d)*HW + h0*WW + wpix] = fmaxf(val, 0.f);
  }
}

// ---------------------------------------------------------------------------
// q/k/v projections (tiny GEMV per pixel)
// ---------------------------------------------------------------------------
__global__ __launch_bounds__(256) void k_qkv(
    const float* __restrict__ xin, const float* __restrict__ wq,
    const float* __restrict__ wk, const float* __restrict__ wv,
    float* __restrict__ qo, float* __restrict__ ko, float* __restrict__ vo)
{
  __shared__ float wl[320];
  int tid = threadIdx.x;
  for (int i = tid; i < 320; i += 256) {
    float v;
    if (i < 32)       v = wq[i];
    else if (i < 64)  v = wk[i - 32];
    else              v = wv[i - 64];
    wl[i] = v;
  }
  __syncthreads();
  int idx = blockIdx.x * 256 + tid;
  int w = idx & (WW-1);
  int h = (idx >> 7) & (HH-1);
  int d = (idx >> 14) & (DD-1);
  int b = idx >> 18;
  float xc[16];
#pragma unroll
  for (int ci = 0; ci < 16; ++ci)
    xc[ci] = xin[((b*CC + ci)*DD + d)*HW + h*WW + w];
  int pb = b*DD + d;
#pragma unroll
  for (int c = 0; c < 2; ++c) {
    float s0 = 0.f, s1 = 0.f;
#pragma unroll
    for (int ci = 0; ci < 16; ++ci) {
      s0 += wl[c*16 + ci] * xc[ci];
      s1 += wl[32 + c*16 + ci] * xc[ci];
    }
    qo[(pb*2 + c)*HW + h*WW + w] = s0;
    ko[(pb*2 + c)*HW + h*WW + w] = s1;
  }
#pragma unroll
  for (int c = 0; c < 16; ++c) {
    float s = 0.f;
#pragma unroll
    for (int ci = 0; ci < 16; ++ci) s += wl[64 + c*16 + ci] * xc[ci];
    vo[(pb*16 + c)*HW + h*WW + w] = s;
  }
}

// ---------------------------------------------------------------------------
// Row kernel: joint softmax (eH | eW) + oW GEMM (M=w 128, N=c 16, K=v' 128)
// writes xout = xin + (gamma/S)*oW ; stores unnormalized aH (bf16) + gamma/S
// ---------------------------------------------------------------------------
__global__ __launch_bounds__(256) void k_attn_row(
    const float* __restrict__ xin, const float* __restrict__ qw,
    const float* __restrict__ karr, const float* __restrict__ vw,
    const float* __restrict__ gamma_p,
    __bf16* __restrict__ aHw, float* __restrict__ sinvw,
    float* __restrict__ xout)
{
  __shared__ float qr[2][128], kr[2][128];
  __shared__ __bf16 vB[16][132];
  __shared__ __bf16 aWl[128][132];
  __shared__ float red[256];
  __shared__ float sinv[128];

  int tid = threadIdx.x;
  int h0 = blockIdx.x % HH;
  int d  = (blockIdx.x / HH) % DD;
  int b  = blockIdx.x / (HH*DD);
  int pb = b*DD + d;

  {
    int c = tid >> 7, w = tid & 127;
    qr[c][w] = qw[(pb*2 + c)*HW + h0*WW + w];
    kr[c][w] = karr[(pb*2 + c)*HW + h0*WW + w];
  }
  for (int i = tid; i < 16*128; i += 256) {
    int c = i >> 7, w = i & 127;
    vB[c][w] = (__bf16)vw[(pb*16 + c)*HW + h0*WW + w];
  }
  __syncthreads();

  int half = tid >> 7, w = tid & 127;
  float q0 = qr[0][w], q1 = qr[1][w];
  float mx = -3.0e38f;
  if (half == 0) {
    for (int j = 0; j < HH; ++j) {
      if (j == h0) continue;
      float e = q0 * karr[(pb*2 + 0)*HW + j*WW + w]
              + q1 * karr[(pb*2 + 1)*HW + j*WW + w];
      mx = fmaxf(mx, e);
    }
  } else {
    for (int vv = 0; vv < WW; ++vv)
      mx = fmaxf(mx, q0 * kr[0][vv] + q1 * kr[1][vv]);
  }
  red[tid] = mx;
  __syncthreads();
  float M = fmaxf(red[w], red[w + 128]);
  __syncthreads();

  float s = 0.f;
  if (half == 0) {
    __bf16* dst = aHw + (((size_t)pb*WW + w)*HH + h0)*HH;
    for (int j = 0; j < HH; ++j) {
      float p = 0.f;
      if (j != h0) {
        float e = q0 * karr[(pb*2 + 0)*HW + j*WW + w]
                + q1 * karr[(pb*2 + 1)*HW + j*WW + w];
        p = __expf(e - M);
      }
      s += p;
      dst[j] = (__bf16)p;
    }
  } else {
    for (int vv = 0; vv < WW; ++vv) {
      float p = __expf(q0 * kr[0][vv] + q1 * kr[1][vv] - M);
      s += p;
      aWl[w][vv] = (__bf16)p;
    }
  }
  red[tid] = s;
  __syncthreads();
  if (half == 0) {
    float S = red[w] + red[w + 128];
    float si = gamma_p[0] / S;
    sinv[w] = si;
    sinvw[(pb*HH + h0)*WW + w] = si;
  }
  __syncthreads();

  int lane = tid & 31, wave = tid >> 5;
  int m = lane & 15, kset = lane >> 4;
  int wbase = wave * 16;
  v8f acc; for (int i = 0; i < 8; ++i) acc[i] = 0.f;
  for (int kb = 0; kb < 4; ++kb) {
    FragAB A, Bf;
#pragma unroll
    for (int t = 0; t < 8; ++t) {
      int k = kb*32 + (t < 4 ? (kset*8 + 2*t) : (16 + kset*8 + 2*(t-4)));
      A.u[t]  = *(const unsigned*)&aWl[wbase + m][k];
      Bf.u[t] = *(const unsigned*)&vB[m][k];
    }
    acc = wmma_bf16(A, Bf, acc);
  }
#pragma unroll
  for (int r = 0; r < 8; ++r) {
    int wpix = wbase + r + 8*kset;
    size_t addr = ((size_t)(b*CC + m)*DD + d)*HW + (size_t)h0*WW + wpix;
    xout[addr] = xin[addr] + sinv[wpix] * acc[r];
  }
}

// ---------------------------------------------------------------------------
// Column kernel: oH GEMM (M=h 128, N=c 16, K=j 128), RMW add (gamma/S)*oH
// ---------------------------------------------------------------------------
__global__ __launch_bounds__(256) void k_attn_col(
    const float* __restrict__ vw, const __bf16* __restrict__ aHw,
    const float* __restrict__ sinvw, float* __restrict__ xout)
{
  __shared__ __bf16 vC[16][132];
  __shared__ __bf16 aHl[128][132];
  __shared__ float sb[128];

  int tid = threadIdx.x;
  int w0 = blockIdx.x % WW;
  int d  = (blockIdx.x / WW) % DD;
  int b  = blockIdx.x / (WW*DD);
  int pb = b*DD + d;

  for (int i = tid; i < 16*128; i += 256) {
    int c = i >> 7, j = i & 127;
    vC[c][j] = (__bf16)vw[(pb*16 + c)*HW + j*WW + w0];
  }
  const unsigned* src = (const unsigned*)(aHw + ((size_t)pb*WW + w0)*HH*HH);
  for (int i = tid; i < 128*64; i += 256) {
    int h = i >> 6, jj = i & 63;
    ((unsigned*)&aHl[h][0])[jj] = src[h*64 + jj];
  }
  if (tid < 128) sb[tid] = sinvw[(pb*HH + tid)*WW + w0];
  __syncthreads();

  int lane = tid & 31, wave = tid >> 5;
  int m = lane & 15, kset = lane >> 4;
  int hbase = wave * 16;
  v8f acc; for (int i = 0; i < 8; ++i) acc[i] = 0.f;
  for (int kb = 0; kb < 4; ++kb) {
    FragAB A, Bf;
#pragma unroll
    for (int t = 0; t < 8; ++t) {
      int k = kb*32 + (t < 4 ? (kset*8 + 2*t) : (16 + kset*8 + 2*(t-4)));
      A.u[t]  = *(const unsigned*)&aHl[hbase + m][k];
      Bf.u[t] = *(const unsigned*)&vC[m][k];
    }
    acc = wmma_bf16(A, Bf, acc);
  }
#pragma unroll
  for (int r = 0; r < 8; ++r) {
    int hh = hbase + r + 8*kset;
    size_t addr = ((size_t)(b*CC + m)*DD + d)*HW + (size_t)hh*WW + w0;
    xout[addr] += sb[hh] * acc[r];
  }
}

// ---------------------------------------------------------------------------
// 3x3x3 conv (16->64) + BN + gate nonlin : M=128, N=64, K=448 (padded 432)
// ---------------------------------------------------------------------------
__global__ __launch_bounds__(256) void k_gateconv(
    const float* __restrict__ xin, const __bf16* __restrict__ BwGate,
    const float* __restrict__ biasGate, float* __restrict__ gates)
{
  __shared__ __bf16 xp[3][3][132][16];
  __shared__ __bf16 Bg[64*448];
  int tid = threadIdx.x;
  int h0 = blockIdx.x % HH;
  int d  = (blockIdx.x / HH) % DD;
  int b  = blockIdx.x / (HH*DD);

  for (int i = tid; i < 3*3*130*16; i += 256) {
    int kd = i / 6240, r = i % 6240;
    int kh = r / 2080; int r2 = r % 2080;
    int wsi = r2 >> 4, ci = r2 & 15;
    int dd = d - 1 + kd, hh = h0 - 1 + kh, ww = wsi - 1;
    float val = 0.f;
    if (dd >= 0 && dd < DD && hh >= 0 && hh < HH && ww >= 0 && ww < WW)
      val = xin[((b*CC + ci)*DD + dd)*HW + hh*WW + ww];
    xp[kd][kh][wsi][ci] = (__bf16)val;
  }
  for (int i = tid; i < 64*448/2; i += 256)
    ((unsigned*)Bg)[i] = ((const unsigned*)BwGate)[i];
  __syncthreads();

  int lane = tid & 31, wave = tid >> 5;
  int m = lane & 15, kset = lane >> 4;
  int wbase = wave * 16;
  v8f acc[4];
  for (int nt = 0; nt < 4; ++nt)
    for (int i = 0; i < 8; ++i) acc[nt][i] = 0.f;

  for (int kb = 0; kb < 14; ++kb) {
    FragAB A;
#pragma unroll
    for (int t = 0; t < 8; ++t) {
      int k = kb*32 + (t < 4 ? (kset*8 + 2*t) : (16 + kset*8 + 2*(t-4)));
      int ci = k & 15, idx = k >> 4;
      unsigned dv = 0u;
      if (idx < 27) {
        int kd = idx / 9, rr = idx % 9, kh = rr / 3, kwid = rr % 3;
        dv = *(const unsigned*)&xp[kd][kh][wbase + m + kwid][ci];
      }
      A.u[t] = dv;
    }
#pragma unroll
    for (int nt = 0; nt < 4; ++nt) {
      FragAB Bf;
#pragma unroll
      for (int t = 0; t < 8; ++t)
        Bf.u[t] = *(const unsigned*)&Bg[(nt*16 + m)*448 + kb*32 + (t < 4 ? (kset*8 + 2*t) : (16 + kset*8 + 2*(t-4)))];
      acc[nt] = wmma_bf16(A, Bf, acc[nt]);
    }
  }
#pragma unroll
  for (int nt = 0; nt < 4; ++nt) {
    float bias = biasGate[nt*16 + m];
#pragma unroll
    for (int r = 0; r < 8; ++r) {
      int wpix = wbase + r + 8*kset;
      float val = acc[nt][r] + bias;
      float a = (nt == 0 || nt == 3) ? tanhf(val)
                                     : 1.f / (1.f + __expf(-val));
      gates[(((size_t)(b*4 + nt)*16 + m)*DD + d)*HW + (size_t)h0*WW + wpix] = a;
    }
  }
}

// ---------------------------------------------------------------------------
// SRU-style scan over D
// ---------------------------------------------------------------------------
__global__ __launch_bounds__(256) void k_scan(
    const float* __restrict__ gates, float* __restrict__ out)
{
  int idx = blockIdx.x * 256 + threadIdx.x;
  int w  = idx & 127;
  int h  = (idx >> 7) & 127;
  int co = (idx >> 14) & 15;
  int b  = idx >> 18;
  size_t sp = (size_t)h*WW + w;
  const float* gw = gates + ((size_t)(b*4 + 0)*16 + co)*DD*HW;
  const float* gf = gates + ((size_t)(b*4 + 1)*16 + co)*DD*HW;
  const float* gr = gates + ((size_t)(b*4 + 2)*16 + co)*DD*HW;
  const float* gx = gates + ((size_t)(b*4 + 3)*16 + co)*DD*HW;
  float Ct = 0.f;
  for (int dd = 0; dd < DD; ++dd) {
    size_t o = (size_t)dd*HW + sp;
    float ft = gf[o], rt = gr[o], xx = gx[o];
    if (dd == 0) Ct = 1.f - ft;
    else { float wx = gw[o]; Ct = ft*Ct + (1.f - ft)*wx; }
    out[((size_t)(b*CC + co)*DD + dd)*HW + sp] = rt*Ct + (1.f - rt)*xx;
  }
}

// ---------------------------------------------------------------------------
extern "C" void kernel_launch(void* const* d_in, const int* in_sizes, int n_in,
                              void* d_out, int out_size, void* d_ws, size_t ws_size,
                              hipStream_t stream) {
  (void)in_sizes; (void)n_in; (void)out_size; (void)ws_size;
  const float* x     = (const float*)d_in[0];
  const float* w_pre = (const float*)d_in[1];
  const float* bpg   = (const float*)d_in[2];
  const float* bpb   = (const float*)d_in[3];
  const float* bpm   = (const float*)d_in[4];
  const float* bpv   = (const float*)d_in[5];
  const float* w_q   = (const float*)d_in[6];
  const float* w_k   = (const float*)d_in[7];
  const float* w_v   = (const float*)d_in[8];
  const float* gamma = (const float*)d_in[9];
  const float* w_g   = (const float*)d_in[10];
  const float* bgg   = (const float*)d_in[11];
  const float* bgb   = (const float*)d_in[12];
  const float* bgm   = (const float*)d_in[13];
  const float* bgv   = (const float*)d_in[14];

  float* ws = (float*)d_ws;
  const size_t NX = (size_t)BB*CC*DD*HW;
  float* x1 = ws;        ws += NX;
  float* x2 = ws;        ws += NX;
  float* qw = ws;        ws += (size_t)BB*DD*2*HW;
  float* kw = ws;        ws += (size_t)BB*DD*2*HW;
  float* vw = ws;        ws += NX;
  float* sinvw = ws;     ws += (size_t)BB*DD*HW;
  float* biasPre = ws;   ws += 16;
  float* biasGate = ws;  ws += 64;
  float* gates = ws;     ws += (size_t)BB*64*DD*HW;
  __bf16* aHw    = (__bf16*)ws;
  __bf16* BwPre  = aHw + (size_t)BB*DD*WW*HH*HH;
  __bf16* BwGate = BwPre + 16*160;

  dim3 blk(256);
  k_prep<<<1, blk, 0, stream>>>(w_pre, bpg, bpb, bpm, bpv,
                                w_g, bgg, bgb, bgm, bgv,
                                BwPre, biasPre, BwGate, biasGate);
  k_preconv<<<BB*DD*HH, blk, 0, stream>>>(x, BwPre, biasPre, x1);

  // attention pass 1: x1 -> x2
  k_qkv<<<(BB*DD*HW)/256, blk, 0, stream>>>(x1, w_q, w_k, w_v, qw, kw, vw);
  k_attn_row<<<BB*DD*HH, blk, 0, stream>>>(x1, qw, kw, vw, gamma, aHw, sinvw, x2);
  k_attn_col<<<BB*DD*WW, blk, 0, stream>>>(vw, aHw, sinvw, x2);
  // attention pass 2: x2 -> x1
  k_qkv<<<(BB*DD*HW)/256, blk, 0, stream>>>(x2, w_q, w_k, w_v, qw, kw, vw);
  k_attn_row<<<BB*DD*HH, blk, 0, stream>>>(x2, qw, kw, vw, gamma, aHw, sinvw, x1);
  k_attn_col<<<BB*DD*WW, blk, 0, stream>>>(vw, aHw, sinvw, x1);

  k_gateconv<<<BB*DD*HH, blk, 0, stream>>>(x1, BwGate, biasGate, gates);
  k_scan<<<(BB*CC*HW)/256, blk, 0, stream>>>(gates, (float*)d_out);
}
